// MTL_Net_TF_48481590837424
// MI455X (gfx1250) — matrix operand order repaired and checked by reading, not compiled, not run on hardware
//
#include <hip/hip_runtime.h>
#include <hip/hip_bf16.h>

typedef __attribute__((ext_vector_type(16))) _Float16 v16h;
typedef __attribute__((ext_vector_type(8)))  _Float16 v8h;
typedef __attribute__((ext_vector_type(8)))  float    v8f;

// ---------------------------------------------------------------------------
// Prep kernels
// ---------------------------------------------------------------------------

__global__ void f32_to_f16_kernel(const float* __restrict__ src,
                                  _Float16* __restrict__ dst, int n) {
    int i = blockIdx.x * blockDim.x + threadIdx.x;
    if (i < n) dst[i] = (_Float16)src[i];
}

// Pack a [K,N] f32 matrix (optionally reading it transposed: src is [N,K])
// into WMMA-B fragment layout for v_wmma_f32_16x16x32_f16:
// tiles of 32(K) x 16(N); within a tile, 32 lanes x 16 contiguous halfs:
//   lane L: n = nt*16 + (L&15), k-block = kt*32 + (L>=16 ? 16 : 0),
//   element e (0..15): k = k-block + e   (VGPR v=e/2 holds k=2v,2v+1)
__global__ void pack_b_kernel(const float* __restrict__ src,
                              _Float16* __restrict__ dst,
                              int K, int N, int transpose, int total) {
    int idx = blockIdx.x * blockDim.x + threadIdx.x;
    if (idx >= total) return;
    int within = idx & 511;
    int tile   = idx >> 9;
    int Ntiles = N >> 4;
    int kt = tile / Ntiles;
    int nt = tile - kt * Ntiles;
    int lane = within >> 4;
    int e    = within & 15;
    int k = kt * 32 + ((lane >> 4) << 4) + e;
    int n = nt * 16 + (lane & 15);
    float v = transpose ? src[(size_t)n * K + k] : src[(size_t)k * N + n];
    dst[idx] = (_Float16)v;
}

// Build S2_t[r0,r1] = sum_{r2} S[r0,r1,r2] * U2[t,r2] directly in packed
// B-fragment layout with k = r1, n = r0.  dst: [T][Ktiles][Ntiles][512]
__global__ void pack_s2_kernel(const float* __restrict__ S,
                               const float* __restrict__ U2,
                               _Float16* __restrict__ dst,
                               int R, int R2, int total) {
    int idx = blockIdx.x * blockDim.x + threadIdx.x;
    if (idx >= total) return;
    int within = idx & 511;
    int tile   = idx >> 9;
    int Ntiles = R >> 4;
    int Ktiles = R >> 5;
    int perT   = Ktiles * Ntiles;
    int t   = tile / perT;
    int rem = tile - t * perT;
    int kt = rem / Ntiles;
    int nt = rem - kt * Ntiles;
    int lane = within >> 4;
    int e    = within & 15;
    int k = kt * 32 + ((lane >> 4) << 4) + e;   // r1
    int n = nt * 16 + (lane & 15);              // r0
    const float* srow = S + ((size_t)n * R + k) * R2;
    const float* u    = U2 + (size_t)t * R2;
    float acc = 0.f;
    for (int r2 = 0; r2 < R2; ++r2) acc += srow[r2] * u[r2];
    dst[idx] = (_Float16)acc;
}

// ---------------------------------------------------------------------------
// Fragment loaders
// ---------------------------------------------------------------------------

// A fragment: lane holds K = hi*16 + {0..15} (contiguous)
__device__ __forceinline__ v16h load_a_frag(const _Float16* p) {
    v8h x0 = *(const v8h*)(p);
    v8h x1 = *(const v8h*)(p + 8);
    return __builtin_shufflevector(x0, x1, 0,1,2,3,4,5,6,7,8,9,10,11,12,13,14,15);
}

// B fragment (packed layout): 16 contiguous halfs per lane
__device__ __forceinline__ v16h load_b_frag(const _Float16* p) {
    v8h x0 = *(const v8h*)(p);
    v8h x1 = *(const v8h*)(p + 8);
    return __builtin_shufflevector(x0, x1, 0,1,2,3,4,5,6,7,8,9,10,11,12,13,14,15);
}

// ---------------------------------------------------------------------------
// WMMA GEMM: C[z] = A[z] (MxK, f16 row-major) * Bp[z] (packed fragments)
// optional +bias[n*T+z], optional ReLU; writes f16 row-major and/or f32
// strided [m][n][z] (final output layout).
// One wave (32 threads) computes a 32(M) x 64(N) tile: 8 accumulators,
// 2 A fragments x 4 B fragments per 32-wide k-step -> 12 b128 loads per
// 8 WMMAs (1.5 loads/WMMA), 8 independent WMMA chains per wave.
// ---------------------------------------------------------------------------
__global__ void gemm_wmma_kernel(const _Float16* __restrict__ A,  long long sA,
                                 const _Float16* __restrict__ Bp, long long sB,
                                 _Float16* __restrict__ C16, long long sC,
                                 float* __restrict__ C32,
                                 const float* __restrict__ bias,
                                 int M, int N, int K, int Tt, int relu) {
    int ntb = blockIdx.x * 4;        // first of 4 consecutive 16-wide N tiles
    int mtb = blockIdx.y * 2;        // first of 2 consecutive 16-tall M tiles
    int z   = blockIdx.z;
    const _Float16* Az = A  + (size_t)z * sA;
    const _Float16* Bz = Bp + (size_t)z * sB;

    int lane = threadIdx.x;          // 0..31
    int lo   = lane & 15;
    int hi   = lane >> 4;            // half-wave select
    int Ktiles = K >> 5;
    int Ntiles = N >> 4;

    v8f accA[4] = {}, accB[4] = {};

    // A fragments for the two 16-row blocks
    const _Float16* arow0 = Az + (size_t)(mtb * 16 + lo) * K + hi * 16;
    const _Float16* arow1 = arow0 + (size_t)16 * K;
    // B fragments for the 4 subtiles are consecutive 512-half blocks
    const _Float16* bfrag = Bz + (size_t)ntb * 512 + lane * 16;
    size_t bstride = (size_t)Ntiles * 512;

    for (int kt = 0; kt < Ktiles; ++kt) {
        v16h a0 = load_a_frag(arow0);
        v16h a1 = load_a_frag(arow1);
#pragma unroll
        for (int j = 0; j < 4; ++j) {
            v16h b = load_b_frag(bfrag + j * 512);
            accA[j] = __builtin_amdgcn_wmma_f32_16x16x32_f16(
                          false, a0, false, b, (short)0, accA[j], false, false);
            accB[j] = __builtin_amdgcn_wmma_f32_16x16x32_f16(
                          false, a1, false, b, (short)0, accB[j], false, false);
        }
        arow0 += 32;
        arow1 += 32;
        bfrag += bstride;
    }

    // C layout: lanes 0-15 hold rows mbase..mbase+7, lanes 16-31 rows +8..+15
    int mbase0 = mtb * 16 + hi * 8;
    int mbase1 = mbase0 + 16;
    _Float16* Cz = C16 ? (C16 + (size_t)z * sC) : nullptr;

#pragma unroll
    for (int j = 0; j < 4; ++j) {
        int n = (ntb + j) * 16 + lo;
        float bv = bias ? bias[(size_t)n * Tt + z] : 0.f;
        if (Cz) {
#pragma unroll
            for (int v = 0; v < 8; ++v) {
                float val0 = accA[j][v] + bv;
                float val1 = accB[j][v] + bv;
                if (relu) {
                    val0 = val0 > 0.f ? val0 : 0.f;
                    val1 = val1 > 0.f ? val1 : 0.f;
                }
                Cz[(size_t)(mbase0 + v) * N + n] = (_Float16)val0;
                Cz[(size_t)(mbase1 + v) * N + n] = (_Float16)val1;
            }
        }
        if (C32) {
#pragma unroll
            for (int v = 0; v < 8; ++v) {
                float val0 = accA[j][v] + bv;
                float val1 = accB[j][v] + bv;
                if (relu) {
                    val0 = val0 > 0.f ? val0 : 0.f;
                    val1 = val1 > 0.f ? val1 : 0.f;
                }
                C32[((size_t)(mbase0 + v) * N + n) * Tt + z] = val0;
                C32[((size_t)(mbase1 + v) * N + n) * Tt + z] = val1;
            }
        }
    }
}

// ---------------------------------------------------------------------------
// Host launcher
// ---------------------------------------------------------------------------

static inline void run_gemm(hipStream_t s,
                            const _Float16* A, long long sA,
                            const _Float16* Bp, long long sB,
                            _Float16* C16, long long sC,
                            float* C32, const float* bias,
                            int M, int N, int K, int Tt, int nb, int relu) {
    dim3 grid(N / 64, M / 32, nb), block(32);
    gemm_wmma_kernel<<<grid, block, 0, s>>>(A, sA, Bp, sB, C16, sC, C32, bias,
                                            M, N, K, Tt, relu);
}

extern "C" void kernel_launch(void* const* d_in, const int* in_sizes, int n_in,
                              void* d_out, int out_size, void* d_ws, size_t ws_size,
                              hipStream_t stream) {
    (void)in_sizes; (void)n_in; (void)out_size; (void)ws_size;

    const int Bsz = 512, NIN = 1024, SZ0 = 2048, SZ1 = 2048, SZ2 = 1024;
    const int T = 8, R = 256, R2 = 8;

    const float* x    = (const float*)d_in[0];
    const float* lS0  = (const float*)d_in[1];
    const float* lU00 = (const float*)d_in[2];
    const float* lU10 = (const float*)d_in[3];
    const float* lU20 = (const float*)d_in[4];
    const float* lb0  = (const float*)d_in[5];
    const float* lS1  = (const float*)d_in[6];
    const float* lU01 = (const float*)d_in[7];
    const float* lU11 = (const float*)d_in[8];
    const float* lU21 = (const float*)d_in[9];
    const float* lb1  = (const float*)d_in[10];
    const float* lS2  = (const float*)d_in[11];
    const float* lU02 = (const float*)d_in[12];
    const float* lU12 = (const float*)d_in[13];
    const float* lU22 = (const float*)d_in[14];
    const float* lb2  = (const float*)d_in[15];
    float* out = (float*)d_out;

    // Workspace carve-up (halfs), 256B-aligned regions.
    _Float16* ws = (_Float16*)d_ws;
    size_t off = 0;
    auto alloc = [&](size_t n) {
        _Float16* p = ws + off;
        off += (n + 127) & ~(size_t)127;
        return p;
    };
    _Float16* xh   = alloc((size_t)Bsz * NIN);
    _Float16* U1p0 = alloc((size_t)(NIN / 32) * (R / 16) * 512);    // 262144
    _Float16* U1p1 = alloc((size_t)(SZ0 / 32) * (R / 16) * 512);    // 524288
    _Float16* U1p2 = alloc((size_t)(SZ1 / 32) * (R / 16) * 512);    // 524288
    _Float16* S2p0 = alloc((size_t)T * (R / 32) * (R / 16) * 512);  // 524288
    _Float16* S2p1 = alloc((size_t)T * (R / 32) * (R / 16) * 512);
    _Float16* S2p2 = alloc((size_t)T * (R / 32) * (R / 16) * 512);
    _Float16* U0p0 = alloc((size_t)(R / 32) * (SZ0 / 16) * 512);    // 524288
    _Float16* U0p1 = alloc((size_t)(R / 32) * (SZ1 / 16) * 512);
    _Float16* U0p2 = alloc((size_t)(R / 32) * (SZ2 / 16) * 512);    // 262144
    _Float16* G    = alloc((size_t)T * Bsz * R);                    // 1M halfs
    _Float16* H    = alloc((size_t)T * Bsz * R);
    _Float16* hA   = alloc((size_t)T * Bsz * SZ0);                  // 8M halfs
    _Float16* hB   = alloc((size_t)T * Bsz * SZ1);

    auto blocks = [](int n) { return (n + 255) / 256; };

    // ---- Prep: f16 conversion + B-fragment packing (recomputed every call)
    {
        int n = Bsz * NIN;
        f32_to_f16_kernel<<<blocks(n), 256, 0, stream>>>(x, xh, n);
    }
    {
        int tot = (NIN / 32) * (R / 16) * 512;
        pack_b_kernel<<<blocks(tot), 256, 0, stream>>>(lU10, U1p0, NIN, R, 0, tot);
        tot = (SZ0 / 32) * (R / 16) * 512;
        pack_b_kernel<<<blocks(tot), 256, 0, stream>>>(lU11, U1p1, SZ0, R, 0, tot);
        pack_b_kernel<<<blocks(tot), 256, 0, stream>>>(lU12, U1p2, SZ1, R, 0, tot);
        // U0 used transposed: B[k=r0][n=j] = U0[j][r0]
        tot = (R / 32) * (SZ0 / 16) * 512;
        pack_b_kernel<<<blocks(tot), 256, 0, stream>>>(lU00, U0p0, R, SZ0, 1, tot);
        pack_b_kernel<<<blocks(tot), 256, 0, stream>>>(lU01, U0p1, R, SZ1, 1, tot);
        tot = (R / 32) * (SZ2 / 16) * 512;
        pack_b_kernel<<<blocks(tot), 256, 0, stream>>>(lU02, U0p2, R, SZ2, 1, tot);
        // S2_t core contractions
        tot = T * (R / 32) * (R / 16) * 512;
        pack_s2_kernel<<<blocks(tot), 256, 0, stream>>>(lS0, lU20, S2p0, R, R2, tot);
        pack_s2_kernel<<<blocks(tot), 256, 0, stream>>>(lS1, lU21, S2p1, R, R2, tot);
        pack_s2_kernel<<<blocks(tot), 256, 0, stream>>>(lS2, lU22, S2p2, R, R2, tot);
    }

    const long long sG = (long long)Bsz * R;                    // 131072
    const long long sS = (long long)(R / 32) * (R / 16) * 512;  // 65536
    const long long sH20 = (long long)Bsz * SZ0;                // 1048576

    // ---- Layer 0 (input broadcast over tasks -> step 1 is task-independent)
    run_gemm(stream, xh, 0, U1p0, 0, G, 0, nullptr, nullptr,
             Bsz, R, NIN, T, 1, 0);                         // G0 = x * U1
    run_gemm(stream, G, 0, S2p0, sS, H, sG, nullptr, nullptr,
             Bsz, R, R, T, T, 0);                           // H_t = G0 * S2_t
    run_gemm(stream, H, sG, U0p0, 0, hA, sH20, nullptr, lb0,
             Bsz, SZ0, R, T, T, 1);                         // h1_t = relu(H_t*U0^T + b)

    // ---- Layer 1
    run_gemm(stream, hA, sH20, U1p1, 0, G, sG, nullptr, nullptr,
             Bsz, R, SZ0, T, T, 0);
    run_gemm(stream, G, sG, S2p1, sS, H, sG, nullptr, nullptr,
             Bsz, R, R, T, T, 0);
    run_gemm(stream, H, sG, U0p1, 0, hB, sH20, nullptr, lb1,
             Bsz, SZ1, R, T, T, 1);

    // ---- Layer 2 (no ReLU; f32 output in [b][j][t] layout)
    run_gemm(stream, hB, sH20, U1p2, 0, G, sG, nullptr, nullptr,
             Bsz, R, SZ1, T, T, 0);
    run_gemm(stream, G, sG, S2p2, sS, H, sG, nullptr, nullptr,
             Bsz, R, R, T, T, 0);
    run_gemm(stream, H, sG, U0p2, 0, nullptr, 0, out, lb2,
             Bsz, SZ2, R, T, T, 0);
}
